// MAEGINDecoder_9749575762319
// MI455X (gfx1250) — matrix operand order repaired and checked by reading, not compiled, not run on hardware
//
#include <hip/hip_runtime.h>
#include <hip/hip_bf16.h>

#define N_NODES 100000
#define N_EDGES 1600000
#define HIDDEN  128
#define VOCAB   2048

typedef __attribute__((ext_vector_type(16))) __bf16 v16bf;
typedef __attribute__((ext_vector_type(8)))  __bf16 v8bf;
typedef __attribute__((ext_vector_type(8)))  float  v8f;

union ABfrag { v16bf v; v8bf h2[2]; };

// round-to-nearest-even f32 -> bf16 bits
__device__ __forceinline__ unsigned short f2bf(float f) {
    unsigned int u = __float_as_uint(f);
    u += 0x7FFFu + ((u >> 16) & 1u);
    return (unsigned short)(u >> 16);
}

// ---------------------------------------------------------------------------
// 1) Transpose + downconvert both weight matrices to column-major bf16.
//    wtb[n*128 + k] = bf16(W_trn[k*128 + n])   (128x128)
//    wpb[n*128 + k] = bf16(W_prd[k*2048 + n])  (2048x128)
// ---------------------------------------------------------------------------
__global__ void conv_w_kernel(const float* __restrict__ Wt, const float* __restrict__ Wp,
                              unsigned short* __restrict__ wtb, unsigned short* __restrict__ wpb) {
    int i = blockIdx.x * blockDim.x + threadIdx.x;
    if (i < HIDDEN * HIDDEN) {
        int n = i >> 7, k = i & 127;
        wtb[i] = f2bf(Wt[k * HIDDEN + n]);
    } else if (i < HIDDEN * HIDDEN + VOCAB * HIDDEN) {
        int j = i - HIDDEN * HIDDEN;
        int n = j >> 7, k = j & 127;
        wpb[j] = f2bf(Wp[k * VOCAB + n]);
    }
}

// ---------------------------------------------------------------------------
// 2) h = x  (self term of GIN with eps=0)
// ---------------------------------------------------------------------------
__global__ void copy_f4_kernel(const float4* __restrict__ src, float4* __restrict__ dst, int n4) {
    int i = blockIdx.x * blockDim.x + threadIdx.x;
    if (i < n4) dst[i] = src[i];
}

// ---------------------------------------------------------------------------
// 3) Edge scatter: one wave per edge, each lane handles 4 floats of the row.
//    h[dst] += x[src]; accumulator (51.2 MB) is L2-resident -> hw fp32 atomics.
// ---------------------------------------------------------------------------
__global__ void scatter_kernel(const float* __restrict__ x, const int* __restrict__ ei,
                               float* __restrict__ h) {
    int wid  = threadIdx.x >> 5;
    int lane = threadIdx.x & 31;
    long e = (long)blockIdx.x * 8 + wid;
    if (e >= N_EDGES) return;
    int src = ei[e];
    int dst = ei[N_EDGES + e];
    const float4 v = *(const float4*)(x + (long)src * HIDDEN + lane * 4);
    float* hp = h + (long)dst * HIDDEN + lane * 4;
    unsafeAtomicAdd(hp + 0, v.x);
    unsafeAtomicAdd(hp + 1, v.y);
    unsafeAtomicAdd(hp + 2, v.z);
    unsafeAtomicAdd(hp + 3, v.w);
}

// ---------------------------------------------------------------------------
// 4) fp32 -> bf16 downconvert of the aggregated activations
// ---------------------------------------------------------------------------
__global__ void cvt_bf_kernel(const float* __restrict__ s, unsigned short* __restrict__ d, long n) {
    long i = ((long)blockIdx.x * blockDim.x + threadIdx.x) * 4;
    if (i < n) {
        float4 v = *(const float4*)(s + i);
        ushort4 o;
        o.x = f2bf(v.x); o.y = f2bf(v.y); o.z = f2bf(v.z); o.w = f2bf(v.w);
        *(ushort4*)(d + i) = o;
    }
}

// ---------------------------------------------------------------------------
// Shared 16x16 tile GEMM core: C(16x16,f32) += A(16x128,bf16) * B(128x16,bf16)
// A row-major (row stride 128), B column-major (col stride 128).
// Fragment layouts (this code's own k order, same for A and B):
//   A lane L: M = L%16; elem e -> K = 32*kk + 16*(L/16) + e   (contiguous)
//   B lane L: N = L%16; elem e -> K = 32*kk + 16*(L/16) + e   (contiguous!)
// ---------------------------------------------------------------------------
__device__ __forceinline__ void gemm_tile_k128(const unsigned short* __restrict__ A,
                                               const unsigned short* __restrict__ Bt,
                                               int rowBase, int colBase, v8f& c) {
    int lane = threadIdx.x & 31;
    int n16  = lane & 15;
    int hf   = lane >> 4;
    const __bf16* arow = (const __bf16*)A  + (long)(rowBase + n16) * HIDDEN;
    const __bf16* bcol = (const __bf16*)Bt + (long)(colBase + n16) * HIDDEN;
#pragma unroll
    for (int kk = 0; kk < 4; ++kk) {
        ABfrag a;
        const __bf16* ap = arow + kk * 32 + 16 * hf;
        a.h2[0] = *(const v8bf*)(ap);
        a.h2[1] = *(const v8bf*)(ap + 8);
        v16bf b = *(const v16bf*)(bcol + kk * 32 + 16 * hf);
        c = __builtin_amdgcn_wmma_f32_16x16x32_bf16(false, a.v, false, b,
                                                    (short)0, c, false, false);
    }
}

// ---------------------------------------------------------------------------
// 5) t = bf16(h @ W_trn + b_trn); 8 waves/block = 8 column tiles per 16 rows
// ---------------------------------------------------------------------------
__global__ void gemm1_kernel(const unsigned short* __restrict__ hb,
                             const unsigned short* __restrict__ wtb,
                             const float* __restrict__ b_trn,
                             unsigned short* __restrict__ tb) {
    int wave = threadIdx.x >> 5;
    int lane = threadIdx.x & 31;
    int rowBase = blockIdx.x * 16;
    int colBase = wave * 16;
    v8f c = {};
    gemm_tile_k128(hb, wtb, rowBase, colBase, c);
    int n16 = lane & 15, hf = lane >> 4;
    float bias = b_trn[colBase + n16];
#pragma unroll
    for (int r = 0; r < 8; ++r) {
        tb[(long)(rowBase + r + 8 * hf) * HIDDEN + colBase + n16] = f2bf(c[r] + bias);
    }
}

// ---------------------------------------------------------------------------
// 6) logits = t @ W_prd + b_prd ; 6250 row tiles x 128 col tiles, 8 tiles/block
// ---------------------------------------------------------------------------
__global__ void gemm2_kernel(const unsigned short* __restrict__ tb,
                             const unsigned short* __restrict__ wpb,
                             const float* __restrict__ b_prd,
                             float* __restrict__ out) {
    int wave = threadIdx.x >> 5;
    int lane = threadIdx.x & 31;
    int rowBase = (blockIdx.x >> 4) * 16;
    int colBase = ((blockIdx.x & 15) * 8 + wave) * 16;
    v8f c = {};
    gemm_tile_k128(tb, wpb, rowBase, colBase, c);
    int n16 = lane & 15, hf = lane >> 4;
    float bias = b_prd[colBase + n16];
#pragma unroll
    for (int r = 0; r < 8; ++r) {
        out[(long)(rowBase + r + 8 * hf) * VOCAB + colBase + n16] = c[r] + bias;
    }
}

extern "C" void kernel_launch(void* const* d_in, const int* in_sizes, int n_in,
                              void* d_out, int out_size, void* d_ws, size_t ws_size,
                              hipStream_t stream) {
    const float* x  = (const float*)d_in[0];
    const int*   ei = (const int*)d_in[1];
    const float* Wt = (const float*)d_in[2];
    const float* bt = (const float*)d_in[3];
    const float* Wp = (const float*)d_in[4];
    const float* bp = (const float*)d_in[5];
    (void)in_sizes; (void)n_in; (void)out_size; (void)ws_size;

    char* ws = (char*)d_ws;
    float*          h   = (float*)ws;                          // 51,200,000 B
    unsigned short* hb  = (unsigned short*)(ws +  51200000);   // 25,600,000 B
    unsigned short* tb  = (unsigned short*)(ws +  76800000);   // 25,600,000 B
    unsigned short* wtb = (unsigned short*)(ws + 102400000);   //     32,768 B
    unsigned short* wpb = (unsigned short*)(ws + 102432768);   //    524,288 B

    const int nW = HIDDEN * HIDDEN + VOCAB * HIDDEN;           // 278,528
    conv_w_kernel<<<(nW + 255) / 256, 256, 0, stream>>>(Wt, Wp, wtb, wpb);

    const int n4 = N_NODES * HIDDEN / 4;                       // 3,200,000
    copy_f4_kernel<<<(n4 + 255) / 256, 256, 0, stream>>>((const float4*)x, (float4*)h, n4);

    scatter_kernel<<<N_EDGES / 8, 256, 0, stream>>>(x, ei, h); // 200,000 blocks

    cvt_bf_kernel<<<(n4 + 255) / 256, 256, 0, stream>>>(h, hb, (long)N_NODES * HIDDEN);

    gemm1_kernel<<<N_NODES / 16, 256, 0, stream>>>(hb, wtb, bt, tb);          // 6,250 blocks
    gemm2_kernel<<<(N_NODES / 16) * 16, 256, 0, stream>>>(tb, wpb, bp, (float*)d_out); // 100,000 blocks
}
